// Attention_60533269070083
// MI455X (gfx1250) — compile-verified, matrix operand order checked by reading, not run on hardware
//
#include <hip/hip_runtime.h>
#include <hip/hip_bf16.h>

// ---------------------------------------------------------------------------
// Attention block for MI455X (gfx1250, wave32, WMMA 16x16x32 f16 -> f32 acc)
//   B=2, N=2048, C=1024, H=16, D=64, SCALE=0.125
// Pipeline: f32->f16 convert | QKV GEMM (+Vt scatter) | flash attention | proj
// Round 4: no peeled tail (unconditional next-k fragment loads; the 64B
//          last-iteration over-read stays inside d_ws by region ordering),
//          __launch_bounds__(256,1) to stop scratch spills of fragments.
// ---------------------------------------------------------------------------

typedef __attribute__((ext_vector_type(16))) _Float16 v16h;
typedef __attribute__((ext_vector_type(8)))  _Float16 v8h;
typedef __attribute__((ext_vector_type(4)))  _Float16 v4h;
typedef __attribute__((ext_vector_type(8)))  float    v8f;

#define ATT_B 2
#define ATT_N 2048
#define ATT_C 1024
#define ATT_H 16
#define ATT_D 64
#define ATT_3C 3072
#define ATT_ROWS 4096           // B*N
#define ATT_SCALE 0.125f        // 64^-0.5

static __device__ __forceinline__ v8f zero8f() {
    v8f z;
#pragma unroll
    for (int i = 0; i < 8; ++i) z[i] = 0.0f;
    return z;
}

// Build a 16-half WMMA fragment from two contiguous 8-half (16B) chunks.
// Per the CDNA5 16-bit A/B layout, lane (half = lane>>4) holds
// K = half*8 + {0..7} in elements 0..7 and K = 16 + half*8 + {0..7} in 8..15.
static __device__ __forceinline__ v16h load_frag(const _Float16* rowk0, int half) {
    v8h c0 = *(const v8h*)(rowk0 + half * 8);
    v8h c1 = *(const v8h*)(rowk0 + 16 + half * 8);
    v16h r;
#pragma unroll
    for (int i = 0; i < 8; ++i) { r[i] = c0[i]; r[8 + i] = c1[i]; }
    return r;
}

static __device__ __forceinline__ v8f wmma_f16(v16h a, v16h b, v8f c) {
    return __builtin_amdgcn_wmma_f32_16x16x32_f16(false, a, false, b,
                                                  (short)0, c, false, false);
}

// ---------------------------------------------------------------------------
// Kernel 0: fp32 -> f16 convert, 4 elements/thread (n must be multiple of 4)
// ---------------------------------------------------------------------------
__global__ void att_f2h(const float* __restrict__ s, _Float16* __restrict__ d, int n) {
    const int stride = gridDim.x * blockDim.x * 4;
    for (int i = (blockIdx.x * blockDim.x + threadIdx.x) * 4; i < n; i += stride) {
        const float4 f = *(const float4*)(s + i);
        v4h o;
        o[0] = (_Float16)f.x; o[1] = (_Float16)f.y;
        o[2] = (_Float16)f.z; o[3] = (_Float16)f.w;
        *(v4h*)(d + i) = o;
    }
}

// ---------------------------------------------------------------------------
// Software-pipelined 32x64 GEMM core, running pointers, NO peeled tail:
// next-k fragments are loaded unconditionally every iteration; the final
// iteration over-reads 64B past the row end, which by workspace region
// ordering is always still inside d_ws (values are discarded).
// A is [M,K] row-major f16; W is [NC,K] row-major f16 (B = W^T).
// ---------------------------------------------------------------------------
static __device__ __forceinline__ void gemm_core_32x64(
    const _Float16* __restrict__ A, const _Float16* __restrict__ W,
    int K, int m0, int n0, int lane, v8f acc[2][4])
{
    const int half = lane >> 4, ln = lane & 15;
    const _Float16* ar0 = A + (size_t)(m0 + ln) * K;
    const _Float16* ar1 = ar0 + (size_t)16 * K;
    const _Float16* br0 = W + (size_t)(n0 + ln) * K;
    const _Float16* br1 = br0 + (size_t)16 * K;
    const _Float16* br2 = br0 + (size_t)32 * K;
    const _Float16* br3 = br0 + (size_t)48 * K;

    v16h a0 = load_frag(ar0, half), a1 = load_frag(ar1, half);
    v16h b0 = load_frag(br0, half), b1 = load_frag(br1, half);
    v16h b2 = load_frag(br2, half), b3 = load_frag(br3, half);

    const int iters = K / 32;
    for (int it = 0; it < iters; ++it) {
        // unconditional prefetch of k+32 fragments (benign over-read last iter)
        v16h na0 = load_frag(ar0 + 32, half);
        v16h na1 = load_frag(ar1 + 32, half);
        v16h nb0 = load_frag(br0 + 32, half);
        v16h nb1 = load_frag(br1 + 32, half);
        v16h nb2 = load_frag(br2 + 32, half);
        v16h nb3 = load_frag(br3 + 32, half);

        acc[0][0] = wmma_f16(a0, b0, acc[0][0]);
        acc[1][0] = wmma_f16(a1, b0, acc[1][0]);
        acc[0][1] = wmma_f16(a0, b1, acc[0][1]);
        acc[1][1] = wmma_f16(a1, b1, acc[1][1]);
        acc[0][2] = wmma_f16(a0, b2, acc[0][2]);
        acc[1][2] = wmma_f16(a1, b2, acc[1][2]);
        acc[0][3] = wmma_f16(a0, b3, acc[0][3]);
        acc[1][3] = wmma_f16(a1, b3, acc[1][3]);

        a0 = na0; a1 = na1;
        b0 = nb0; b1 = nb1; b2 = nb2; b3 = nb3;
        ar0 += 32; ar1 += 32;
        br0 += 32; br1 += 32; br2 += 32; br3 += 32;
    }
}

// ---------------------------------------------------------------------------
// Kernel 1: QKV GEMM.  C[4096,3072] = Xh[4096,1024] @ Wqkv^T (+bias).
// Q columns (<1024) get (val + q_bias)*SCALE; K columns get no bias;
// V columns (>=2048) get v_bias and are ALSO scattered into Vt[b,h,d,key].
// One wave computes a 32x64 tile.
// ---------------------------------------------------------------------------
__global__ void __launch_bounds__(256, 1)
att_gemm_qkv(const _Float16* __restrict__ A,   // [4096,1024]
             const _Float16* __restrict__ W,   // [3072,1024]
             const float* __restrict__ q_bias, // [1024]
             const float* __restrict__ v_bias, // [1024]
             _Float16* __restrict__ Cout,      // [4096,3072]
             _Float16* __restrict__ Vt)        // [B*H*64, 2048]
{
    const int lane = threadIdx.x & 31;
    const int wave = threadIdx.x >> 5;
    const int wid  = blockIdx.x * 8 + wave;
    const int ngrp = ATT_3C / 64;            // 48 column groups
    const int m0 = (wid / ngrp) * 32;        // 128 row tiles of 32
    const int n0 = (wid % ngrp) * 64;
    const int half = lane >> 4, ln = lane & 15;

    v8f acc[2][4];
#pragma unroll
    for (int r = 0; r < 2; ++r)
#pragma unroll
        for (int t = 0; t < 4; ++t) acc[r][t] = zero8f();

    gemm_core_32x64(A, W, ATT_C, m0, n0, lane, acc);

#pragma unroll
    for (int t = 0; t < 4; ++t) {
        const int col = n0 + t * 16 + ln;
        float bias = 0.0f, scale = 1.0f;
        if (col < ATT_C)       { bias = q_bias[col];          scale = ATT_SCALE; }
        else if (col >= 2048)  { bias = v_bias[col - 2048]; }
#pragma unroll
        for (int r = 0; r < 2; ++r)
#pragma unroll
            for (int v = 0; v < 8; ++v) {
                const int row = m0 + r * 16 + v + 8 * half;
                const float val = (acc[r][t][v] + bias) * scale;
                Cout[(size_t)row * ATT_3C + col] = (_Float16)val;
                if (col >= 2048) {
                    const int c = col - 2048;
                    const int h = c >> 6, d = c & 63;
                    const int b = row >> 11;           // row / 2048
                    const int key = row & 2047;
                    Vt[(((size_t)b * ATT_H + h) * ATT_D + d) * ATT_N + key] = (_Float16)val;
                }
            }
    }
}

// ---------------------------------------------------------------------------
// Kernel 2: flash attention. One wave owns a 16-query tile of one (b,h).
// Running pointers (constant-step) for K rows, Vt runs and bias rows.
// Per 32-key block: S = Q K^T (4 WMMA) + rel_pos_bias, online softmax
// (shfl-width-16 row reductions), P via LDS -> A fragment, O += P V (4 WMMA).
// ---------------------------------------------------------------------------
__global__ void __launch_bounds__(256, 1)
att_flash(const _Float16* __restrict__ QKV, // [4096, 3072]
          const _Float16* __restrict__ Vt,  // [B*H*64, 2048]
          const float* __restrict__ bias,   // [H, N, N]
          _Float16* __restrict__ Oh)        // [4096, 1024]
{
    __shared__ __align__(16) _Float16 lds[8][16 * 40];      // per-wave 16x32 P tile (pad 40)

    const int lane = threadIdx.x & 31;
    const int wave = threadIdx.x >> 5;
    const int wid  = blockIdx.x * 8 + wave;
    const int rt = wid & 127;               // 128 row tiles
    const int bh = wid >> 7;
    const int h = bh & 15, b = bh >> 4;
    const int m0 = rt * 16;
    const int half = lane >> 4, ln = lane & 15;

    // Q fragments (already scaled by SCALE in the QKV GEMM)
    const _Float16* qrow = QKV + (size_t)(b * ATT_N + m0 + ln) * ATT_3C + h * ATT_D;
    const v16h aq0 = load_frag(qrow,      half);
    const v16h aq1 = load_frag(qrow + 32, half);

    v8f o[4];
#pragma unroll
    for (int t = 0; t < 4; ++t) o[t] = zero8f();
    float mrow[8], lrow[8];
#pragma unroll
    for (int v = 0; v < 8; ++v) { mrow[v] = -1e30f; lrow[v] = 0.0f; }

    // Running pointers, advanced by constants each 32-key block.
    const _Float16* kr0 = QKV + (size_t)(b * ATT_N + ln) * ATT_3C + ATT_C + h * ATT_D;
    const _Float16* kr1 = kr0 + (size_t)16 * ATT_3C;
    const _Float16* vr  = Vt + (((size_t)b * ATT_H + h) * ATT_D + ln) * ATT_N;
    const float*    bp  = bias + ((size_t)h * ATT_N + m0) * ATT_N + ln;

    for (int kb = 0; kb < ATT_N; kb += 32) {
        // Prefetch next block's K rows / Vt run (gfx1250 global_prefetch_b8;
        // +32-key offsets stay inside the workspace for the final block).
        __builtin_prefetch(kr0 + (size_t)32 * ATT_3C, 0, 3);
        __builtin_prefetch(kr1 + (size_t)32 * ATT_3C, 0, 3);
        __builtin_prefetch(vr + 32, 0, 3);
        // bias prefetch clamped (input buffer ends exactly at the last row)
        const float* bpf = (kb + 32 < ATT_N) ? (bp + 32) : bp;
#pragma unroll
        for (int v = 0; v < 8; ++v)
            __builtin_prefetch(bpf + (size_t)(v + 8 * half) * ATT_N, 0, 3);

        // ---- S = Q K^T for 32 keys (two 16x16 tiles) ----
        v16h bk00 = load_frag(kr0,      half);
        v16h bk01 = load_frag(kr0 + 32, half);
        v16h bk10 = load_frag(kr1,      half);
        v16h bk11 = load_frag(kr1 + 32, half);
        v8f s[2];
        s[0] = zero8f(); s[1] = zero8f();
        s[0] = wmma_f16(aq0, bk00, s[0]);
        s[0] = wmma_f16(aq1, bk01, s[0]);
        s[1] = wmma_f16(aq0, bk10, s[1]);
        s[1] = wmma_f16(aq1, bk11, s[1]);

        // ---- + rel_pos_bias (the real HBM stream: 256 MB total) ----
#pragma unroll
        for (int t = 0; t < 2; ++t)
#pragma unroll
            for (int v = 0; v < 8; ++v)
                s[t][v] += bp[(size_t)(v + 8 * half) * ATT_N + t * 16];

        // ---- online softmax (rows live in 16-lane groups) ----
        float alpha[8];
#pragma unroll
        for (int v = 0; v < 8; ++v) {
            float mx = fmaxf(s[0][v], s[1][v]);
#pragma unroll
            for (int off = 8; off >= 1; off >>= 1)
                mx = fmaxf(mx, __shfl_xor(mx, off, 16));
            const float mnew = fmaxf(mrow[v], mx);
            alpha[v] = __expf(mrow[v] - mnew);
            mrow[v] = mnew;
            const float p0 = __expf(s[0][v] - mnew);
            const float p1 = __expf(s[1][v] - mnew);
            s[0][v] = p0; s[1][v] = p1;
            float rs = p0 + p1;
#pragma unroll
            for (int off = 8; off >= 1; off >>= 1)
                rs += __shfl_xor(rs, off, 16);
            lrow[v] = lrow[v] * alpha[v] + rs;
        }
#pragma unroll
        for (int t = 0; t < 4; ++t)
#pragma unroll
            for (int v = 0; v < 8; ++v) o[t][v] *= alpha[v];

        // ---- P (C layout) -> LDS -> A fragment (cross-lane transpose) ----
        __syncthreads();
#pragma unroll
        for (int t = 0; t < 2; ++t)
#pragma unroll
            for (int v = 0; v < 8; ++v)
                lds[wave][(v + 8 * half) * 40 + t * 16 + ln] = (_Float16)s[t][v];
        __syncthreads();
        const v16h ap = load_frag(&lds[wave][ln * 40], half);

        // ---- O += P V  (Vt gives contiguous key runs per d column) ----
        v16h bv0 = load_frag(vr,                        half);
        v16h bv1 = load_frag(vr + (size_t)16 * ATT_N,   half);
        v16h bv2 = load_frag(vr + (size_t)32 * ATT_N,   half);
        v16h bv3 = load_frag(vr + (size_t)48 * ATT_N,   half);
        o[0] = wmma_f16(ap, bv0, o[0]);
        o[1] = wmma_f16(ap, bv1, o[1]);
        o[2] = wmma_f16(ap, bv2, o[2]);
        o[3] = wmma_f16(ap, bv3, o[3]);

        kr0 += (size_t)32 * ATT_3C;
        kr1 += (size_t)32 * ATT_3C;
        vr  += 32;
        bp  += 32;
    }

    // ---- normalize and write [b, n, h*64+d] ----
#pragma unroll
    for (int t = 0; t < 4; ++t)
#pragma unroll
        for (int v = 0; v < 8; ++v) {
            const int row = m0 + v + 8 * half;
            const int col = h * ATT_D + t * 16 + ln;
            Oh[(size_t)(b * ATT_N + row) * ATT_C + col] = (_Float16)(o[t][v] / lrow[v]);
        }
}

// ---------------------------------------------------------------------------
// Kernel 3: output projection. Out[4096,1024] = Ah @ proj_w^T + proj_b (fp32)
// ---------------------------------------------------------------------------
__global__ void __launch_bounds__(256, 1)
att_gemm_proj(const _Float16* __restrict__ A,  // [4096,1024]
              const _Float16* __restrict__ W,  // [1024,1024]
              const float* __restrict__ pb,    // [1024]
              float* __restrict__ Out)         // [4096,1024]
{
    const int lane = threadIdx.x & 31;
    const int wave = threadIdx.x >> 5;
    const int wid  = blockIdx.x * 8 + wave;
    const int ngrp = ATT_C / 64;             // 16 column groups
    const int m0 = (wid / ngrp) * 32;        // 128 row tiles of 32
    const int n0 = (wid % ngrp) * 64;
    const int half = lane >> 4, ln = lane & 15;

    v8f acc[2][4];
#pragma unroll
    for (int r = 0; r < 2; ++r)
#pragma unroll
        for (int t = 0; t < 4; ++t) acc[r][t] = zero8f();

    gemm_core_32x64(A, W, ATT_C, m0, n0, lane, acc);

#pragma unroll
    for (int t = 0; t < 4; ++t) {
        const int col = n0 + t * 16 + ln;
        const float bias = pb[col];
#pragma unroll
        for (int r = 0; r < 2; ++r)
#pragma unroll
            for (int v = 0; v < 8; ++v) {
                const int row = m0 + r * 16 + v + 8 * half;
                Out[(size_t)row * ATT_C + col] = acc[r][t][v] + bias;
            }
    }
}

// ---------------------------------------------------------------------------
// Launch.  Workspace region order guarantees every GEMM A/W operand region
// has a successor region inside d_ws, so the benign +32-half over-reads of
// the pipelined core never leave the workspace:
//   XH -> WQKV -> WPROJ -> ATTH -> QKVH -> VTH
// (QKV/Vt are only read with exact addresses or prefetches.)
// ---------------------------------------------------------------------------
extern "C" void kernel_launch(void* const* d_in, const int* in_sizes, int n_in,
                              void* d_out, int out_size, void* d_ws, size_t ws_size,
                              hipStream_t stream) {
    const float* x        = (const float*)d_in[0];   // [2,2048,1024]
    const float* rel_bias = (const float*)d_in[1];   // [16,2048,2048]
    const float* qkv_w    = (const float*)d_in[2];   // [3072,1024]
    const float* q_bias   = (const float*)d_in[3];   // [1024]
    const float* v_bias   = (const float*)d_in[4];   // [1024]
    const float* proj_w   = (const float*)d_in[5];   // [1024,1024]
    const float* proj_b   = (const float*)d_in[6];   // [1024]
    float* out = (float*)d_out;

    _Float16* ws = (_Float16*)d_ws;
    const size_t XH    = 0;                                 // x f16        [4096,1024]
    const size_t WQKV  = XH    + (size_t)ATT_ROWS * ATT_C;  // qkv_w f16    [3072,1024]
    const size_t WPROJ = WQKV  + (size_t)ATT_3C * ATT_C;    // proj_w f16   [1024,1024]
    const size_t ATTH  = WPROJ + (size_t)ATT_C * ATT_C;     // attn out f16 [4096,1024]
    const size_t QKVH  = ATTH  + (size_t)ATT_ROWS * ATT_C;  // qkv f16      [4096,3072]
    const size_t VTH   = QKVH  + (size_t)ATT_ROWS * ATT_3C; // V^T f16      [2048,2048]

    att_f2h<<<1024, 256, 0, stream>>>(x,      ws + XH,    ATT_ROWS * ATT_C);
    att_f2h<<<1024, 256, 0, stream>>>(qkv_w,  ws + WQKV,  ATT_3C * ATT_C);
    att_f2h<<<512,  256, 0, stream>>>(proj_w, ws + WPROJ, ATT_C * ATT_C);

    // 4096/32 row tiles * 48 col groups = 6144 waves -> 768 blocks of 8 waves
    att_gemm_qkv<<<768, 256, 0, stream>>>(ws + XH, ws + WQKV, q_bias, v_bias,
                                          ws + QKVH, ws + VTH);

    // 2*16*128 = 4096 waves -> 512 blocks
    att_flash<<<512, 256, 0, stream>>>(ws + QKVH, ws + VTH, rel_bias, ws + ATTH);

    // 128 row tiles * 16 col groups = 2048 waves -> 256 blocks
    att_gemm_proj<<<256, 256, 0, stream>>>(ws + ATTH, ws + WPROJ, proj_b, out);

    (void)in_sizes; (void)n_in; (void)out_size; (void)ws_size;
}
